// GGUFLinear_87729001988215
// MI455X (gfx1250) — compile-verified, matrix operand order checked by reading, not run on hardware
//
#include <hip/hip_runtime.h>
#include <stdint.h>

// ---------------------------------------------------------------------------
// GGUF Q4_0 dequant + GEMM for MI455X (gfx1250, wave32, WMMA).
//   out[M,N] = x[M,K] * W[N,K]^T,  M=8192, N=4096, K=4096
//   W stored as Q4_0: block b = (n*128 + k/32): 16 packed bytes + f16 scale.
//
// v2: software-pipelined global prefetch (loads for tile k+1 issued before
//     computing tile k), and v_perm_b32 + packed-f16 dequant:
//       f16bits(0x6400 | q) == 1024+q exactly; (1024+q)-1032 == q-8 exactly;
//       then one v_pk_mul_f16 by the block scale (single rounding, matches
//       the f32 reference semantics).
// ---------------------------------------------------------------------------

#define IN_F   4096
#define OUT_F  4096
#define BLK_M  128
#define BLK_N  128
#define K_STEP 64
#define LDS_K  72   // 64 + 8 halves pad: row stride 144B = 36 dwords (bank-conflict free)

typedef __attribute__((ext_vector_type(2)))  _Float16 v2h;
typedef __attribute__((ext_vector_type(4)))  _Float16 v4h;
typedef __attribute__((ext_vector_type(8)))  _Float16 v8h;
typedef __attribute__((ext_vector_type(16))) _Float16 v16h;
typedef __attribute__((ext_vector_type(8)))  float    v8f;

// Build a 32-byte WMMA fragment from two 16B-aligned LDS loads (ds_load_b128).
__device__ __forceinline__ v16h load_frag(const _Float16* p) {
    v8h lo = *(const v8h*)(p);
    v8h hi = *(const v8h*)(p + 8);
    return __builtin_shufflevector(lo, hi, 0, 1, 2, 3, 4, 5, 6, 7,
                                           8, 9, 10, 11, 12, 13, 14, 15);
}

// Expand two nibble-bytes of `nibs` (bytes b0,b1, values 0..15) into packed
// f16 pair (q-8)*d using one v_perm_b32 + v_pk_add_f16 + v_pk_mul_f16.
__device__ __forceinline__ v2h dq_pair(unsigned nibs, unsigned sel, v2h dd) {
    const unsigned p = __builtin_amdgcn_perm(0x64646464u, nibs, sel);
    const v2h kc = {(_Float16)1032.0f, (_Float16)1032.0f};
    v2h q = __builtin_bit_cast(v2h, p);   // {1024+q0, 1024+q1}
    return (q - kc) * dd;                 // exact sub, then scale
}

__global__ __launch_bounds__(256)
void GGUFLinear_q40_wmma_kernel(const float* __restrict__ x,
                                const uint8_t* __restrict__ q_packed,
                                const _Float16* __restrict__ q_scales,
                                float* __restrict__ out) {
    __shared__ _Float16 xs[BLK_M * LDS_K];   // activations, f16, [m][k]
    __shared__ _Float16 ws[BLK_N * LDS_K];   // dequantized weights, f16, [n][k]

    const int tid  = threadIdx.x;
    const int lane = tid & 31;
    const int wv   = tid >> 5;          // 8 waves
    const int wM   = wv >> 1;           // 0..3 : 32 rows of M each
    const int wN   = wv & 1;            // 0..1 : 64 cols of N each
    const int lrow  = lane & 15;        // row within 16x16 tile
    const int khalf = (lane >> 4) * 16; // lanes 0-15: K 0..15, lanes 16-31: K 16..31

    const int nbase = blockIdx.x * BLK_N;
    const int mbase = blockIdx.y * BLK_M;

    // x staging decomposition: 16 col-quads x 16 rows per pass, 8 passes
    const int cx = tid & 15;            // col quad (4 floats)
    const int r0 = tid >> 4;            // starting row 0..15

    // W staging decomposition: one Q4_0 block (32 weights) per thread per step
    const int wn = tid >> 1;            // n within tile, 0..127
    const int kb = tid & 1;             // which 32-wide k-block of the 64 k-step

    v8f zero = {};
    v8f acc[2][4];
#pragma unroll
    for (int i = 0; i < 2; ++i)
#pragma unroll
        for (int j = 0; j < 4; ++j) acc[i][j] = zero;

    // ---- software-pipeline prefetch state ----
    float4   xv[8];
    uint4    qraw;
    _Float16 dscale;

    const float*   xrow = &x[(size_t)(mbase + r0) * IN_F + cx * 4];
    const size_t   blk0 = (size_t)(nbase + wn) * (IN_F / 32) + kb;

#define LOAD_TILES(KK)                                                        \
    do {                                                                      \
        _Pragma("unroll")                                                     \
        for (int rr = 0; rr < 8; ++rr)                                        \
            xv[rr] = *(const float4*)&xrow[(size_t)rr * 16 * IN_F + (KK)];    \
        const size_t blk = blk0 + ((KK) >> 5);                                \
        qraw   = *(const uint4*)(q_packed + blk * 16);                        \
        dscale = q_scales[blk];                                               \
    } while (0)

    LOAD_TILES(0);

    for (int kk = 0; kk < IN_F; kk += K_STEP) {
        // ---- drain prefetch: convert + store this tile into LDS ----
#pragma unroll
        for (int rr = 0; rr < 8; ++rr) {
            v4h h;
            h[0] = (_Float16)xv[rr].x; h[1] = (_Float16)xv[rr].y;
            h[2] = (_Float16)xv[rr].z; h[3] = (_Float16)xv[rr].w;
            *(v4h*)&xs[(r0 + rr * 16) * LDS_K + cx * 4] = h;
        }
        {
            const v2h dd = {dscale, dscale};
            _Float16 tmp[32];
            const unsigned q[4] = {qraw.x, qraw.y, qraw.z, qraw.w};
#pragma unroll
            for (int w = 0; w < 4; ++w) {
                const unsigned lo = q[w] & 0x0F0F0F0Fu;         // quants 4w..4w+3
                const unsigned hi = (q[w] >> 4) & 0x0F0F0F0Fu;  // quants 16+4w..
                const v2h rl0 = dq_pair(lo, 0x05010400u, dd);   // bytes 0,1
                const v2h rl1 = dq_pair(lo, 0x05030402u, dd);   // bytes 2,3
                const v2h rh0 = dq_pair(hi, 0x05010400u, dd);
                const v2h rh1 = dq_pair(hi, 0x05030402u, dd);
                tmp[4 * w + 0]      = rl0[0]; tmp[4 * w + 1]      = rl0[1];
                tmp[4 * w + 2]      = rl1[0]; tmp[4 * w + 3]      = rl1[1];
                tmp[16 + 4 * w + 0] = rh0[0]; tmp[16 + 4 * w + 1] = rh0[1];
                tmp[16 + 4 * w + 2] = rh1[0]; tmp[16 + 4 * w + 3] = rh1[1];
            }
            _Float16* dst = &ws[wn * LDS_K + kb * 32];
#pragma unroll
            for (int c = 0; c < 4; ++c) {
                v8h pack;
#pragma unroll
                for (int e = 0; e < 8; ++e) pack[e] = tmp[c * 8 + e];
                *(v8h*)(dst + c * 8) = pack;   // ds_store_b128
            }
        }

        __syncthreads();   // LDS tile visible to all waves

        // ---- issue next tile's global loads; they fly during compute ----
        if (kk + K_STEP < IN_F) LOAD_TILES(kk + K_STEP);

        // ---- compute: 16 WMMAs per wave per K-step ----
#pragma unroll
        for (int ks = 0; ks < K_STEP; ks += 32) {
            v16h a[2], b[4];
#pragma unroll
            for (int i = 0; i < 2; ++i)
                a[i] = load_frag(&xs[(wM * 32 + i * 16 + lrow) * LDS_K + ks + khalf]);
#pragma unroll
            for (int j = 0; j < 4; ++j)
                b[j] = load_frag(&ws[(wN * 64 + j * 16 + lrow) * LDS_K + ks + khalf]);
#pragma unroll
            for (int i = 0; i < 2; ++i)
#pragma unroll
                for (int j = 0; j < 4; ++j)
                    acc[i][j] = __builtin_amdgcn_wmma_f32_16x16x32_f16(
                        /*neg_a=*/false, a[i], /*neg_b=*/false, b[j],
                        /*c_mod=*/(short)0, acc[i][j],
                        /*reuse_a=*/false, /*reuse_b=*/false);
        }

        __syncthreads();   // compute done before next LDS overwrite
    }

    // ---- epilogue: C/D layout -> global f32 ----
    // VGPR r: lanes 0-15 hold M=r, N=lane; lanes 16-31 hold M=r+8, N=lane-16.
    const int nc = lane & 15;
    const int mo = (lane >> 4) * 8;
#pragma unroll
    for (int i = 0; i < 2; ++i) {
#pragma unroll
        for (int j = 0; j < 4; ++j) {
#pragma unroll
            for (int r = 0; r < 8; ++r) {
                const int m = mbase + wM * 32 + i * 16 + mo + r;
                const int n = nbase + wN * 64 + j * 16 + nc;
                out[(size_t)m * OUT_F + n] = acc[i][j][r];
            }
        }
    }
}

extern "C" void kernel_launch(void* const* d_in, const int* in_sizes, int n_in,
                              void* d_out, int out_size, void* d_ws, size_t ws_size,
                              hipStream_t stream) {
    const float*    x  = (const float*)d_in[0];
    const uint8_t*  qp = (const uint8_t*)d_in[1];
    const _Float16* qs = (const _Float16*)d_in[2];
    float*          o  = (float*)d_out;

    const int M = in_sizes[0] / IN_F;            // 8192
    dim3 grid(OUT_F / BLK_N, M / BLK_M);         // (32, 64)
    dim3 block(256);
    GGUFLinear_q40_wmma_kernel<<<grid, block, 0, stream>>>(x, qp, qs, o);
}
